// RPN_4406636446271
// MI455X (gfx1250) — matrix operand order repaired and checked by reading, not compiled, not run on hardware
//
#include <hip/hip_runtime.h>
#include <hip/hip_bf16.h>
#include <math.h>

// ---------------------------------------------------------------------------
// Problem constants
// ---------------------------------------------------------------------------
#define CIN   1024
#define COUT  512
#define HW    64
#define PIX   4096            // 64*64
#define KTOT  9216            // 1024 * 9
#define NA    9
#define NANCH 36864           // 4096 * 9
#define NPAD  65536           // sort padding
#define NPRE  12000
#define NPOST 2000

// Padded input geometry (66x66 bf16 per channel)
#define XPW   66
#define XPCH  4356            // 66*66
#define NXP   (CIN * XPCH)    // 4,460,544 ushorts

// GEMM tiling
#define BK 32
#define BM 128                // out-channel tile
#define BN 128                // pixel tile
#define KSTEPS 288            // KTOT / BK

// d_out layout (floats): locs | scores | rois | anchors
#define OUT_LOCS   0
#define OUT_SCORES 147456
#define OUT_ROIS   221184
#define OUT_ANCH   229184

typedef __attribute__((ext_vector_type(16))) __bf16 v16bf;
typedef __attribute__((ext_vector_type(8)))  float  v8f;

__device__ __forceinline__ unsigned short f2bf(float f) {
  unsigned int u = __float_as_uint(f);
  u += 0x7FFFu + ((u >> 16) & 1u);       // round-to-nearest-even
  return (unsigned short)(u >> 16);
}

// ---------------------------------------------------------------------------
// Prep 1: repack conv weights into tap-major (k' = tap*1024 + c), bf16,
// WMMA-A-fragment-ready layout:
//   wfrag[(tile_m*KSTEPS + kstep)*512 + (m + 16*half)*16 + e]
// (contiguous k order matching B: half=kk>>4, e=kk&15)
// ---------------------------------------------------------------------------
__global__ void prep_w(const float* __restrict__ w,
                       unsigned short* __restrict__ wfrag) {
  int id = blockIdx.x * 256 + threadIdx.x;   // 0 .. 512*9216-1 (exact multiple)
  float v = w[id];                            // coalesced read
  int o   = id / KTOT;
  int k   = id - o * KTOT;
  int c   = k / 9;
  int tap = k - c * 9;
  int kstep = tap * 32 + (c >> 5);
  int kk    = c & 31;
  int half  = kk >> 4;
  int e     = kk & 15;
  int tile_m = o >> 4, m = o & 15;
  wfrag[(tile_m * KSTEPS + kstep) * 512 + (m + 16 * half) * 16 + e] = f2bf(v);
}

// ---------------------------------------------------------------------------
// Prep 2: zero-pad + bf16-convert input: xp[c][66][66]
// ---------------------------------------------------------------------------
__global__ void prep_x(const float* __restrict__ x,
                       unsigned short* __restrict__ xp) {
  int id = blockIdx.x * 256 + threadIdx.x;   // exact multiple of NXP
  int c   = id / XPCH;
  int rem = id - c * XPCH;
  int iy  = rem / XPW;
  int ix  = rem - iy * XPW;
  float v = 0.f;
  if ((unsigned)(iy - 1) < 64u && (unsigned)(ix - 1) < 64u)
    v = x[c * PIX + (iy - 1) * HW + (ix - 1)];
  xp[id] = f2bf(v);
}

// ---------------------------------------------------------------------------
// Kernel 1: 3x3 conv (1024->512, SAME, leaky-relu) as implicit GEMM via WMMA.
// A fragments stream straight from the pre-swizzled global buffer (2x b128
// per fragment, L2-resident); B patches staged through LDS with a fixed
// per-thread address pattern (16 coalesced u16 loads -> 2x ds_store_b128).
// ---------------------------------------------------------------------------
__global__ __launch_bounds__(256) void conv3x3_wmma(
    const unsigned short* __restrict__ xp,     // [1024][66][66] bf16 padded
    const unsigned short* __restrict__ wfrag,  // fragment-ready weights
    const float* __restrict__ bias, float* __restrict__ hbuf) {
  __shared__ __align__(32) unsigned short ldsB[(BN / 16) * 512];

  const int t    = threadIdx.x;
  const int lane = t & 31;
  const int wid  = t >> 5;        // 0..7
  const int wm   = wid >> 2;      // 0..1  (64 outch each)
  const int wn   = wid & 3;       // 0..3  (32 px each)
  const int pBase = blockIdx.x * BN;
  const int oBase = blockIdx.y * BM;

  // ---- fixed per-thread B staging geometry ----
  // thread t stages kk = khalf*16 + i (i=0..15) for pixel pl = t&127
  const int pl    = t & 127;
  const int khalf = t >> 7;                    // 0 or 1
  const int p  = pBase + pl;
  const int py = p >> 6, px = p & 63;
  // B fragment layout (32x16, K x N): lane = n + 16*(k/16), e = k%16
  // -> this thread's 16 staged values are 16 contiguous LDS ushorts
  unsigned short* ldsDst = &ldsB[(pl >> 4) * 512 + ((pl & 15) + 16 * khalf) * 16];

  v8f acc[4][2];
  #pragma unroll
  for (int i = 0; i < 4; i++)
    #pragma unroll
    for (int j = 0; j < 2; j++) {
      v8f z = {0.f, 0.f, 0.f, 0.f, 0.f, 0.f, 0.f, 0.f};
      acc[i][j] = z;
    }

  for (int tap = 0; tap < 9; tap++) {
    const int kyp = tap / 3, kxp = tap - kyp * 3;   // hoisted, 9x only
    const unsigned short* xbase =
        xp + (khalf * 16) * XPCH + (py + kyp) * XPW + (px + kxp);

    for (int cs = 0; cs < 32; cs++) {               // channel chunk: c0 = cs*32
      const int kstep = tap * 32 + cs;
      __syncthreads();
      // ---- stage B: 16 u16 loads (stride = one channel), pack, 2x b128 ----
      const unsigned short* src = xbase + cs * 32 * XPCH;
      unsigned int pk[8];
      #pragma unroll
      for (int i = 0; i < 8; i++) {
        unsigned int lo = src[(2 * i + 0) * XPCH];
        unsigned int hi = src[(2 * i + 1) * XPCH];
        pk[i] = lo | (hi << 16);
      }
      int4 v0; v0.x = pk[0]; v0.y = pk[1]; v0.z = pk[2]; v0.w = pk[3];
      int4 v1; v1.x = pk[4]; v1.y = pk[5]; v1.z = pk[6]; v1.w = pk[7];
      *reinterpret_cast<int4*>(ldsDst)     = v0;
      *reinterpret_cast<int4*>(ldsDst + 8) = v1;
      if (cs < 31) __builtin_prefetch(src + 32 * XPCH, 0, 1);  // global_prefetch
      __syncthreads();

      // ---- A fragments direct from global (L2-resident, pre-swizzled) ----
      v16bf afr[4], bfr[2];
      #pragma unroll
      for (int mt = 0; mt < 4; mt++) {
        int off = (((oBase >> 4) + wm * 4 + mt) * KSTEPS + kstep) * 512 + lane * 16;
        afr[mt] = *reinterpret_cast<const v16bf*>(wfrag + off);
      }
      #pragma unroll
      for (int nt = 0; nt < 2; nt++)
        bfr[nt] = *reinterpret_cast<const v16bf*>(&ldsB[(wn * 2 + nt) * 512 + lane * 16]);

      #pragma unroll
      for (int mt = 0; mt < 4; mt++)
        #pragma unroll
        for (int nt = 0; nt < 2; nt++)
          acc[mt][nt] = __builtin_amdgcn_wmma_f32_16x16x32_bf16(
              false, afr[mt], false, bfr[nt], (short)0, acc[mt][nt], false, false);
    }
  }

  // ---- epilogue: bias + leaky relu, write h[o][p] ----
  const int mrow = (lane >> 4) * 8;   // C/D layout: VGPR v -> row v + 8*(lane/16)
  const int ncol = lane & 15;
  #pragma unroll
  for (int mt = 0; mt < 4; mt++)
    #pragma unroll
    for (int nt = 0; nt < 2; nt++)
      #pragma unroll
      for (int v = 0; v < 8; v++) {
        int o = oBase + wm * 64 + mt * 16 + mrow + v;
        int pp = pBase + wn * 32 + nt * 16 + ncol;
        float val = acc[mt][nt][v] + bias[o];
        val = val > 0.f ? val : 0.01f * val;
        hbuf[o * PIX + pp] = val;
      }
}

// ---------------------------------------------------------------------------
// Kernel 2: fused RPN heads. One thread per pixel. 1x1 convs via LDS-chunked
// dot products, then softmax fg, anchor generation, box decode.
// ---------------------------------------------------------------------------
__global__ __launch_bounds__(256) void rpn_head(
    const float* __restrict__ hbuf,
    const float* __restrict__ sw, const float* __restrict__ sb,
    const float* __restrict__ lw, const float* __restrict__ lb,
    float* __restrict__ out, float* __restrict__ fg, float* __restrict__ boxes) {
  __shared__ float hT[32 * 256];
  __shared__ float wT[54 * 32];      // 36 loc rows then 18 score rows
  const int t     = threadIdx.x;
  const int pBase = blockIdx.x * 256;

  float accL[36], accS[18];
  #pragma unroll
  for (int i = 0; i < 36; i++) accL[i] = 0.f;
  #pragma unroll
  for (int i = 0; i < 18; i++) accS[i] = 0.f;

  for (int cc = 0; cc < COUT; cc += 32) {
    __syncthreads();
    #pragma unroll
    for (int i = 0; i < 32; i++)
      hT[i * 256 + t] = hbuf[(cc + i) * PIX + pBase + t];
    for (int li = t; li < 54 * 32; li += 256) {
      int o = li >> 5, j = li & 31;
      wT[li] = (o < 36) ? lw[o * COUT + cc + j] : sw[(o - 36) * COUT + cc + j];
    }
    __syncthreads();
    for (int j = 0; j < 32; j++) {
      float hv = hT[j * 256 + t];
      #pragma unroll
      for (int o = 0; o < 36; o++) accL[o] += wT[o * 32 + j] * hv;
      #pragma unroll
      for (int o = 0; o < 18; o++) accS[o] += wT[(36 + o) * 32 + j] * hv;
    }
  }

  const int p  = pBase + t;
  const int py = p >> 6, px = p & 63;
  const float ratios[3] = {0.5f, 1.f, 2.f};
  const float scales[3] = {8.f, 16.f, 32.f};

  #pragma unroll
  for (int a = 0; a < NA; a++) {
    float dy = accL[a * 4 + 0] + lb[a * 4 + 0];
    float dx = accL[a * 4 + 1] + lb[a * 4 + 1];
    float dh = accL[a * 4 + 2] + lb[a * 4 + 2];
    float dw = accL[a * 4 + 3] + lb[a * 4 + 3];
    float s0 = accS[a * 2 + 0] + sb[a * 2 + 0];
    float s1 = accS[a * 2 + 1] + sb[a * 2 + 1];
    int ai = p * NA + a;

    out[OUT_LOCS + ai * 4 + 0] = dy;
    out[OUT_LOCS + ai * 4 + 1] = dx;
    out[OUT_LOCS + ai * 4 + 2] = dh;
    out[OUT_LOCS + ai * 4 + 3] = dw;
    out[OUT_SCORES + ai * 2 + 0] = s0;
    out[OUT_SCORES + ai * 2 + 1] = s1;
    fg[ai] = 1.f / (1.f + expf(-(s1 - s0)));      // softmax[:,1]

    float r  = ratios[a / 3], s = scales[a % 3];
    float ha = 16.f * s * sqrtf(r);
    float wa = 16.f * s * sqrtf(1.f / r);
    float cya = py * 16.f + 8.f, cxa = px * 16.f + 8.f;
    out[OUT_ANCH + ai * 4 + 0] = cya - 0.5f * ha;
    out[OUT_ANCH + ai * 4 + 1] = cxa - 0.5f * wa;
    out[OUT_ANCH + ai * 4 + 2] = cya + 0.5f * ha;
    out[OUT_ANCH + ai * 4 + 3] = cxa + 0.5f * wa;

    float cy = dy * ha + cya, cx = dx * wa + cxa;
    float hh = ha * expf(dh), ww = wa * expf(dw);
    boxes[ai * 4 + 0] = cy - 0.5f * hh;
    boxes[ai * 4 + 1] = cx - 0.5f * ww;
    boxes[ai * 4 + 2] = cy + 0.5f * hh;
    boxes[ai * 4 + 3] = cx + 0.5f * ww;
  }
}

// ---------------------------------------------------------------------------
// Kernel 3: sort init (pad keys with -inf, identity values, zero roi slab)
// ---------------------------------------------------------------------------
__global__ void sort_init(const float* __restrict__ fg, float* __restrict__ keys,
                          int* __restrict__ vals, float* __restrict__ rois) {
  int i = blockIdx.x * 256 + threadIdx.x;
  if (i < NPAD) {
    keys[i] = (i < NANCH) ? fg[i] : -3.402823466e38f;
    vals[i] = i;
  }
  if (i < NPOST * 4) rois[i] = 0.f;
}

// ---------------------------------------------------------------------------
// Kernel 4: one bitonic compare-exchange pass (descending by key)
// ---------------------------------------------------------------------------
__global__ void bitonic_pass(float* __restrict__ keys, int* __restrict__ vals,
                             int j, int k) {
  int i   = blockIdx.x * 256 + threadIdx.x;
  int ixj = i ^ j;
  if (ixj > i) {
    float ki = keys[i], kj = keys[ixj];
    bool up = ((i & k) == 0);
    bool sw = up ? (ki < kj) : (ki > kj);    // descending overall
    if (sw) {
      keys[i] = kj; keys[ixj] = ki;
      int vi = vals[i]; vals[i] = vals[ixj]; vals[ixj] = vi;
    }
  }
}

// ---------------------------------------------------------------------------
// Kernel 5: clip + min-size filter + greedy NMS + compaction.
// Single workgroup, boxes/areas/flags in dynamic LDS (~252KB of 320KB WGP LDS)
// ---------------------------------------------------------------------------
__global__ __launch_bounds__(1024) void nms_kernel(
    const int* __restrict__ vals, const float* __restrict__ boxes,
    const int* __restrict__ imghp, const int* __restrict__ imgwp,
    float* __restrict__ rois) {
  extern __shared__ float smem[];
  float* bx   = smem;                                      // NPRE*4
  float* area = smem + NPRE * 4;                           // NPRE
  unsigned char* sup = (unsigned char*)(smem + NPRE * 5);  // NPRE

  const int t = threadIdx.x;
  const float imgh = (float)imghp[0];
  const float imgw = (float)imgwp[0];

  for (int i = t; i < NPRE; i += 1024) {
    int idx  = vals[i];
    float y1 = boxes[idx * 4 + 0], x1 = boxes[idx * 4 + 1];
    float y2 = boxes[idx * 4 + 2], x2 = boxes[idx * 4 + 3];
    y1 = fminf(fmaxf(y1, 0.f), imgh);
    x1 = fminf(fmaxf(x1, 0.f), imgw);
    y2 = fminf(fmaxf(y2, 0.f), imgh);
    x2 = fminf(fmaxf(x2, 0.f), imgw);
    bx[i * 4 + 0] = y1; bx[i * 4 + 1] = x1;
    bx[i * 4 + 2] = y2; bx[i * 4 + 3] = x2;
    area[i] = (y2 - y1) * (x2 - x1);
    bool valid = ((y2 - y1) >= 16.f) && ((x2 - x1) >= 16.f);
    sup[i] = valid ? 0 : 1;
  }
  __syncthreads();

  for (int i = 0; i < NPRE; i++) {
    if (sup[i] == 0) {
      float y1 = bx[i * 4 + 0], x1 = bx[i * 4 + 1];
      float y2 = bx[i * 4 + 2], x2 = bx[i * 4 + 3];
      float ai = area[i];
      for (int j = i + 1 + t; j < NPRE; j += 1024) {
        float yy1 = fmaxf(y1, bx[j * 4 + 0]);
        float xx1 = fmaxf(x1, bx[j * 4 + 1]);
        float yy2 = fminf(y2, bx[j * 4 + 2]);
        float xx2 = fminf(x2, bx[j * 4 + 3]);
        float inter = fmaxf(yy2 - yy1, 0.f) * fmaxf(xx2 - xx1, 0.f);
        float iou = inter / (ai + area[j] - inter + 1e-10f);
        if (iou > 0.7f) sup[j] = 1;
      }
    }
    __syncthreads();
  }

  if (t == 0) {
    int rank = 0;
    for (int i = 0; i < NPRE; i++) {
      if (!sup[i]) {
        if (rank < NPOST) {
          rois[rank * 4 + 0] = bx[i * 4 + 0];
          rois[rank * 4 + 1] = bx[i * 4 + 1];
          rois[rank * 4 + 2] = bx[i * 4 + 2];
          rois[rank * 4 + 3] = bx[i * 4 + 3];
        }
        rank++;
      }
    }
  }
}

// ---------------------------------------------------------------------------
// Host launcher
// ---------------------------------------------------------------------------
extern "C" void kernel_launch(void* const* d_in, const int* in_sizes, int n_in,
                              void* d_out, int out_size, void* d_ws, size_t ws_size,
                              hipStream_t stream) {
  (void)in_sizes; (void)n_in; (void)out_size; (void)ws_size;
  const float* x       = (const float*)d_in[0];
  const float* conv_w  = (const float*)d_in[1];
  const float* conv_b  = (const float*)d_in[2];
  const float* score_w = (const float*)d_in[3];
  const float* score_b = (const float*)d_in[4];
  const float* loc_w   = (const float*)d_in[5];
  const float* loc_b   = (const float*)d_in[6];
  const int*   img_h   = (const int*)d_in[7];
  const int*   img_w   = (const int*)d_in[8];

  float* out = (float*)d_out;
  float* ws  = (float*)d_ws;

  float* hbuf  = ws;                          // 2,097,152 f
  float* fg    = hbuf + (size_t)COUT * PIX;   // 36,864 f
  float* boxes = fg + NANCH;                  // 147,456 f
  float* keys  = boxes + NANCH * 4;           // 65,536 f
  int*   vals  = (int*)(keys + NPAD);         // 65,536 i
  unsigned short* xp    = (unsigned short*)(vals + NPAD);   // 4,460,544 u16 (32B-aligned)
  unsigned short* wfrag = xp + NXP;                          // 4,718,592 u16 (32B-aligned)

  prep_x<<<NXP / 256, 256, 0, stream>>>(x, xp);
  prep_w<<<(COUT * KTOT) / 256, 256, 0, stream>>>(conv_w, wfrag);
  conv3x3_wmma<<<dim3(PIX / BN, COUT / BM), 256, 0, stream>>>(xp, wfrag, conv_b, hbuf);
  rpn_head<<<PIX / 256, 256, 0, stream>>>(hbuf, score_w, score_b, loc_w, loc_b,
                                          out, fg, boxes);
  sort_init<<<NPAD / 256, 256, 0, stream>>>(fg, keys, vals, out + OUT_ROIS);
  for (int k = 2; k <= NPAD; k <<= 1)
    for (int j = k >> 1; j > 0; j >>= 1)
      bitonic_pass<<<NPAD / 256, 256, 0, stream>>>(keys, vals, j, k);
  nms_kernel<<<1, 1024, NPRE * 5 * sizeof(float) + NPRE, stream>>>(
      vals, boxes, img_h, img_w, out + OUT_ROIS);
}
